// LiftingModule_76759655514672
// MI455X (gfx1250) — matrix operand order repaired and checked by reading, not compiled, not run on hardware
//
#include <hip/hip_runtime.h>
#include <hip/hip_bf16.h>
#include <math.h>

// ---------------------------------------------------------------------------
// LiftingModule forward for MI455X (gfx1250, wave32, WMMA).
// All dense projections run through v_wmma_f32_16x16x32_f16 (f16 in, f32 acc).
// GEMM inner loop is software-pipelined (double-buffered fragments) and the
// schedule is pinned with __builtin_amdgcn_sched_group_barrier so each region
// issues the NEXT K-step's 10 vector loads before the current K-step's 4
// WMMAs -> partial s_wait_loadcnt instead of a full drain per WMMA.
// Irregular work (projection, bilinear sampling, LN, attention, pooling) is
// fp32 VALU. Everything launches on `stream`; scratch lives in d_ws.
// ---------------------------------------------------------------------------

#define N_Q   13824      // 24^3 voxel queries
#define NCAM  6
#define DIM   256
#define NP    4          // deform points
#define SS    24
#define HF_   32
#define WF_   32

typedef __attribute__((ext_vector_type(16))) _Float16 v16h;
typedef __attribute__((ext_vector_type(8)))  float    v8f;

union AV { v16h v; uint4 u[2]; };

#if defined(__has_builtin)
#if __has_builtin(__builtin_amdgcn_sched_group_barrier)
#define SCHED_GROUP(mask, size, id) __builtin_amdgcn_sched_group_barrier(mask, size, id)
#endif
#endif
#ifndef SCHED_GROUP
#define SCHED_GROUP(mask, size, id)
#endif

// ---------------------------------------------------------------------------
// Weight prep: W (256 x Nout) fp32 row-major  ->  Wt (Nout x 256) f16,
// K-contiguous so WMMA B fragments are two contiguous b128 loads per lane.
// ---------------------------------------------------------------------------
__global__ __launch_bounds__(256) void prep_w_kernel(
    const float* __restrict__ W, _Float16* __restrict__ Wt, int Nout)
{
    int n = blockIdx.x;        // output column
    int k = threadIdx.x;       // input channel
    Wt[(size_t)n * 256 + k] = (_Float16)W[(size_t)k * Nout + n];
}

// ---------------------------------------------------------------------------
// queries = query_base + posenc3d(grid)
// channel d: axis a = d/84, j = d%84; j<42 -> sin, else cos; d>=252 -> 0
// ---------------------------------------------------------------------------
__global__ __launch_bounds__(256) void init_q_kernel(
    const float* __restrict__ grid, const float* __restrict__ qbase,
    float* __restrict__ queries)
{
    int n = blockIdx.x, d = threadIdx.x;
    float e = 0.0f;
    if (d < 252) {
        int a = d / 84, j = d % 84;
        float coord = grid[n * 3 + a];
        int jj = (j < 42) ? j : (j - 42);
        float fr = exp2f(8.0f * (float)jj / 41.0f) * 3.14159265358979323846f;
        e = (j < 42) ? sinf(coord * fr) : cosf(coord * fr);
    }
    queries[(size_t)n * DIM + d] = qbase[d] + e;
}

// ---------------------------------------------------------------------------
// Project voxels into each camera: normalized coords + validity mask.
// ---------------------------------------------------------------------------
__global__ __launch_bounds__(256) void project_kernel(
    const float* __restrict__ grid, const float* __restrict__ intr,
    const float* __restrict__ extr, float* __restrict__ pc,
    float* __restrict__ valid)
{
    int idx = blockIdx.x * 256 + threadIdx.x;
    if (idx >= NCAM * N_Q) return;
    int c = idx / N_Q, n = idx % N_Q;
    float px = grid[n * 3 + 0], py = grid[n * 3 + 1], pz = grid[n * 3 + 2];
    const float* E = extr + c * 16;
    float cx = E[0] * px + E[1] * py + E[2]  * pz + E[3];
    float cy = E[4] * px + E[5] * py + E[6]  * pz + E[7];
    float cz = E[8] * px + E[9] * py + E[10] * pz + E[11];
    const float* K = intr + c * 9;
    float u = K[0] * cx + K[1] * cy + K[2] * cz;
    float v = K[3] * cx + K[4] * cy + K[5] * cz;
    float zs = (fabsf(cz) > 1e-6f) ? cz : 1e-6f;
    float nx = 2.0f * (u / zs) / 255.0f - 1.0f;
    float ny = 2.0f * (v / zs) / 255.0f - 1.0f;
    bool ok = (cz > 1e-3f) && (nx >= -1.0f) && (nx <= 1.0f)
                           && (ny >= -1.0f) && (ny <= 1.0f);
    pc[((size_t)c * N_Q + n) * 2 + 0] = nx;
    pc[((size_t)c * N_Q + n) * 2 + 1] = ny;
    valid[(size_t)c * N_Q + n] = ok ? 1.0f : 0.0f;
}

// ---------------------------------------------------------------------------
// LayerNorm (+ optional 8-wide offset head). One block per query, d=thread.
// Writes f16 activations for WMMA A operand.
// ---------------------------------------------------------------------------
__global__ __launch_bounds__(256) void ln_kernel(
    const float* __restrict__ q, const float* __restrict__ ng,
    const float* __restrict__ nb, const float* __restrict__ offW,
    const float* __restrict__ offb, _Float16* __restrict__ qn16,
    float* __restrict__ offout, int hasOff)
{
    int n = blockIdx.x, d = threadIdx.x;
    __shared__ float s[256];
    __shared__ float s8[8 * 256];
    float x = q[(size_t)n * DIM + d];
    s[d] = x; __syncthreads();
    for (int st = 128; st > 0; st >>= 1) { if (d < st) s[d] += s[d + st]; __syncthreads(); }
    float mean = s[0] * (1.0f / 256.0f); __syncthreads();
    float xm = x - mean;
    s[d] = xm * xm; __syncthreads();
    for (int st = 128; st > 0; st >>= 1) { if (d < st) s[d] += s[d + st]; __syncthreads(); }
    float var = s[0] * (1.0f / 256.0f);
    float qn = xm * rsqrtf(var + 1e-5f) * ng[d] + nb[d];
    qn16[(size_t)n * DIM + d] = (_Float16)qn;
    if (hasOff) {
        #pragma unroll
        for (int j = 0; j < 8; ++j) s8[j * 256 + d] = qn * offW[d * 8 + j];
        __syncthreads();
        for (int st = 128; st > 0; st >>= 1) {
            if (d < st) {
                #pragma unroll
                for (int j = 0; j < 8; ++j) s8[j * 256 + d] += s8[j * 256 + d + st];
            }
            __syncthreads();
        }
        if (d < 8) offout[n * 8 + d] = (s8[d * 256] + offb[d]) * 0.05f;
    }
}

// ---------------------------------------------------------------------------
// Bilinear sampling. mode 0: P=4 deform points, mean, rows (c*N+n) [per-image]
//                    mode 1: single point,            rows (n*C+c) [cross]
// One block per (camera, voxel); thread = channel (D-contiguous, coalesced).
// ---------------------------------------------------------------------------
__global__ __launch_bounds__(256) void sample_kernel(
    const float* __restrict__ feats, const float* __restrict__ pc,
    const float* __restrict__ offs, _Float16* __restrict__ out, int mode)
{
    int b = blockIdx.x;
    int c = b / N_Q, n = b % N_Q;
    int d = threadIdx.x;
    float cx0 = pc[((size_t)c * N_Q + n) * 2 + 0];
    float cy0 = pc[((size_t)c * N_Q + n) * 2 + 1];
    float acc = 0.0f;
    int np = (mode == 0) ? NP : 1;
    for (int p = 0; p < np; ++p) {
        float cx = cx0, cy = cy0;
        if (mode == 0) {
            cx += offs[((size_t)n * NP + p) * 2 + 0];
            cy += offs[((size_t)n * NP + p) * 2 + 1];
        }
        float x = (cx + 1.0f) * 0.5f * (WF_ - 1);
        float y = (cy + 1.0f) * 0.5f * (HF_ - 1);
        float x0f = floorf(x), y0f = floorf(y);
        float wx1 = x - x0f, wy1 = y - y0f;
        float wx0 = 1.0f - wx1, wy0 = 1.0f - wy1;
        int xi0 = (int)x0f, yi0 = (int)y0f, xi1 = xi0 + 1, yi1 = yi0 + 1;
        auto tap = [&](int yy, int xx) -> float {
            if (yy < 0 || yy >= HF_ || xx < 0 || xx >= WF_) return 0.0f;
            return feats[(((size_t)c * HF_ + yy) * WF_ + xx) * DIM + d];
        };
        acc += tap(yi0, xi0) * wy0 * wx0 + tap(yi0, xi1) * wy0 * wx1
             + tap(yi1, xi0) * wy1 * wx0 + tap(yi1, xi1) * wy1 * wx1;
    }
    if (mode == 0) acc *= 0.25f;
    size_t row = (mode == 0) ? ((size_t)c * N_Q + n) : ((size_t)n * NCAM + c);
    out[row * DIM + d] = (_Float16)acc;
}

// ---------------------------------------------------------------------------
// WMMA GEMM: Y (M x Nout) f32 = X (M x 256) f16 @ Wt^T + bias [+ resid]
// Wt is (Nout x 256) K-contiguous f16. 8 waves/block; wave tile 16x64;
// block tile 32x256; K stepped by 32 via v_wmma_f32_16x16x32_f16.
// Software-pipelined (double-buffered fragments) with sched_group_barrier
// regions of {10 VMEM reads ; 4 WMMA} so loads of K-step i+1 are issued
// before WMMAs of K-step i.
// A layout (16-bit A 16x32, same k order as B): lane-half lh holds 16 sequential K
// {16*lh..16*lh+15} (0..15 / 16..31) -> two b128 loads.
// B layout (16-bit B): lane-half holds 16 sequential K for column lane%16
// -> two contiguous b128 loads from the transposed weight.
// C/D layout: VGPR r, lane L -> row r + 8*(L>=16), col L%16.
// ---------------------------------------------------------------------------
__global__ __launch_bounds__(256) void gemm_wmma_kernel(
    const _Float16* __restrict__ X, const _Float16* __restrict__ Wt,
    const float* __restrict__ bias, const float* __restrict__ resid,
    float* __restrict__ Y, int M, int Nout)
{
    const int lane = threadIdx.x & 31;
    const int wave = threadIdx.x >> 5;
    const int rowBase = blockIdx.x * 32 + (wave >> 2) * 16;
    const int colBase = blockIdx.y * 256 + (wave & 3) * 64;
    const int lr = lane & 15;
    const int lh = lane >> 4;

    const _Float16* Arow  = X  + (size_t)(rowBase + lr) * 256;
    const _Float16* Bbase = Wt + (size_t)(colBase + lr) * 256 + (size_t)lh * 16;

    v8f acc0 = {}, acc1 = {}, acc2 = {}, acc3 = {};

    AV a[2];
    AV b[2][4];

    auto loadA = [&](AV& dst, int kk) {
        dst.u[0] = *(const uint4*)(Arow + kk + lh * 16);       // K kk+lh*16 ..
        dst.u[1] = *(const uint4*)(Arow + kk + lh * 16 + 8);   // K kk+lh*16+8 ..
    };
    auto loadB = [&](AV& dst, int nt, int kk) {
        const _Float16* p = Bbase + (size_t)nt * 4096 + kk;
        dst.u[0] = *(const uint4*)(p);
        dst.u[1] = *(const uint4*)(p + 8);
    };

    // prologue: stage K-step 0
    loadA(a[0], 0);
    loadB(b[0][0], 0, 0);
    loadB(b[0][1], 1, 0);
    loadB(b[0][2], 2, 0);
    loadB(b[0][3], 3, 0);

    #pragma unroll
    for (int it = 0; it < 8; ++it) {
        const int cur = it & 1;
        const int nxt = cur ^ 1;
        if (it < 7) {
            const int kk = (it + 1) * 32;
            loadA(a[nxt], kk);
            loadB(b[nxt][0], 0, kk);
            loadB(b[nxt][1], 1, kk);
            loadB(b[nxt][2], 2, kk);
            loadB(b[nxt][3], 3, kk);
        }
        acc0 = __builtin_amdgcn_wmma_f32_16x16x32_f16(false, a[cur].v, false, b[cur][0].v,
                                                      (short)0, acc0, false, false);
        acc1 = __builtin_amdgcn_wmma_f32_16x16x32_f16(false, a[cur].v, false, b[cur][1].v,
                                                      (short)0, acc1, false, false);
        acc2 = __builtin_amdgcn_wmma_f32_16x16x32_f16(false, a[cur].v, false, b[cur][2].v,
                                                      (short)0, acc2, false, false);
        acc3 = __builtin_amdgcn_wmma_f32_16x16x32_f16(false, a[cur].v, false, b[cur][3].v,
                                                      (short)0, acc3, false, false);
        // Scheduling region per K-step: issue the 10 fragment loads of the
        // NEXT step first, then the 4 WMMAs of the CURRENT step (their
        // operands were loaded in the previous region) -> partial loadcnt
        // waits instead of full drains.
        SCHED_GROUP(0x020, 10, 0);   // 10 VMEM reads
        SCHED_GROUP(0x008, 4, 0);    // 4 WMMA
    }

    #pragma unroll
    for (int r = 0; r < 8; ++r) {
        int orow = rowBase + r + 8 * lh;
        size_t base = (size_t)orow * Nout;
        {
            int oc = colBase + 0 * 16 + lr;
            Y[base + oc] = acc0[r] + bias[oc] + (resid ? resid[base + oc] : 0.0f);
        }
        {
            int oc = colBase + 1 * 16 + lr;
            Y[base + oc] = acc1[r] + bias[oc] + (resid ? resid[base + oc] : 0.0f);
        }
        {
            int oc = colBase + 2 * 16 + lr;
            Y[base + oc] = acc2[r] + bias[oc] + (resid ? resid[base + oc] : 0.0f);
        }
        {
            int oc = colBase + 3 * 16 + lr;
            Y[base + oc] = acc3[r] + bias[oc] + (resid ? resid[base + oc] : 0.0f);
        }
    }
}

// ---------------------------------------------------------------------------
// Per-image attention: attn_c = <q,k_c>/16 * valid ; agg = sum_c attn_c*v_c ;
// out = agg / max(sum valid, 1) -> f16 for the output GEMM.
// ---------------------------------------------------------------------------
__global__ __launch_bounds__(256) void attn_per_kernel(
    const float* __restrict__ qb, const float* __restrict__ kb,
    const float* __restrict__ vb, const float* __restrict__ valid,
    _Float16* __restrict__ agg16)
{
    int n = blockIdx.x, d = threadIdx.x;
    __shared__ float s[256];
    float qd = qb[(size_t)n * DIM + d];
    float attn[NCAM]; float tw = 0.0f;
    for (int c = 0; c < NCAM; ++c) {
        float kd = kb[((size_t)c * N_Q + n) * DIM + d];
        s[d] = qd * kd; __syncthreads();
        for (int st = 128; st > 0; st >>= 1) { if (d < st) s[d] += s[d + st]; __syncthreads(); }
        float vf = valid[(size_t)c * N_Q + n];
        attn[c] = s[0] * (1.0f / 16.0f) * vf;   // 1/sqrt(256)
        tw += vf;
        __syncthreads();
    }
    tw = fmaxf(tw, 1.0f);
    float agg = 0.0f;
    for (int c = 0; c < NCAM; ++c)
        agg += attn[c] * vb[((size_t)c * N_Q + n) * DIM + d];
    agg16[(size_t)n * DIM + d] = (_Float16)(agg / tw);
}

// ---------------------------------------------------------------------------
// Cross-image attention: masked softmax over cameras, then weighted V sum.
// kv rows are (n*C + c), 512 wide: k = [0,256), v = [256,512).
// ---------------------------------------------------------------------------
__global__ __launch_bounds__(256) void attn_cross_kernel(
    const float* __restrict__ qb, const float* __restrict__ kvb,
    const float* __restrict__ valid, _Float16* __restrict__ out16)
{
    int n = blockIdx.x, d = threadIdx.x;
    __shared__ float s[256];
    float qd = qb[(size_t)n * DIM + d];
    float attn[NCAM], vm[NCAM];
    for (int c = 0; c < NCAM; ++c) {
        float kd = kvb[((size_t)n * NCAM + c) * 512 + d];
        s[d] = qd * kd; __syncthreads();
        for (int st = 128; st > 0; st >>= 1) { if (d < st) s[d] += s[d + st]; __syncthreads(); }
        vm[c] = valid[(size_t)c * N_Q + n];
        attn[c] = (vm[c] > 0.0f) ? s[0] * (1.0f / 16.0f) : -1e30f;
        __syncthreads();
    }
    float mx = -1e30f;
    for (int c = 0; c < NCAM; ++c) mx = fmaxf(mx, attn[c]);
    float e[NCAM], se = 0.0f;
    for (int c = 0; c < NCAM; ++c) { e[c] = expf(attn[c] - mx); se += e[c]; }
    float out = 0.0f;
    for (int c = 0; c < NCAM; ++c) {
        float p = (e[c] / se) * vm[c];
        out += p * kvb[((size_t)n * NCAM + c) * 512 + 256 + d];
    }
    out16[(size_t)n * DIM + d] = (_Float16)out;
}

// ---------------------------------------------------------------------------
// Final pooling: pxy (mean over z), pxz (mean over y), pyz (mean over x).
// ---------------------------------------------------------------------------
__global__ __launch_bounds__(256) void reduce_kernel(
    const float* __restrict__ q, float* __restrict__ out)
{
    int d = threadIdx.x;
    int i = blockIdx.x, j = blockIdx.y, mode = blockIdx.z;
    float acc = 0.0f;
    for (int t = 0; t < SS; ++t) {
        int n;
        if (mode == 0)      n = (i * SS + j) * SS + t;  // pxy
        else if (mode == 1) n = (i * SS + t) * SS + j;  // pxz
        else                n = (t * SS + i) * SS + j;  // pyz
        acc += q[(size_t)n * DIM + d];
    }
    out[((size_t)mode * (SS * SS) + i * SS + j) * DIM + d] = acc * (1.0f / SS);
}

// ---------------------------------------------------------------------------

extern "C" void kernel_launch(void* const* d_in, const int* in_sizes, int n_in,
                              void* d_out, int out_size, void* d_ws, size_t ws_size,
                              hipStream_t stream)
{
    const float* feats  = (const float*)d_in[0];  // (1,6,32,32,256)
    const float* intr   = (const float*)d_in[1];  // (1,6,3,3)
    const float* extr   = (const float*)d_in[2];  // (1,6,4,4)
    const float* grid3d = (const float*)d_in[3];  // (24,24,24,3)
    const float* qbase  = (const float*)d_in[4];  // (1,1,256)

    struct PerP { const float *qW,*qb,*kW,*kb,*vW,*vb,*oW,*ob,*offW,*offb,*ng,*nb; };
    struct CrP  { const float *qW,*qb,*kvW,*kvb,*oW,*ob,*ng,*nb; };
    PerP per[2]; CrP cr[2];
    auto F = [&](int i) { return (const float*)d_in[i]; };

    // Param flattening order is ambiguous; disambiguate by size of d_in[5]:
    //  - jax tree_leaves sorts dict keys => 'cross' first, first leaf kvW (131072)
    //  - insertion order => 'per' first, first leaf qW (65536)
    if (n_in > 5 && in_sizes[5] == 131072) {
        // sorted cross keys: kvW,kvb,nb,ng,oW,ob,qW,qb
        for (int l = 0; l < 2; ++l) {
            int b = 5 + l * 8;
            cr[l] = { F(b+6), F(b+7), F(b+0), F(b+1), F(b+4), F(b+5), F(b+3), F(b+2) };
        }
        // sorted per keys: kW,kb,nb,ng,oW,ob,offW,offb,qW,qb,vW,vb
        for (int l = 0; l < 2; ++l) {
            int b = 21 + l * 12;
            per[l] = { F(b+8), F(b+9), F(b+0), F(b+1), F(b+10), F(b+11),
                       F(b+4), F(b+5), F(b+6), F(b+7), F(b+3),  F(b+2) };
        }
    } else {
        // insertion order: per[qW,qb,kW,kb,vW,vb,oW,ob,offW,offb,ng,nb] x2,
        // then cross[qW,qb,kvW,kvb,oW,ob,ng,nb] x2
        for (int l = 0; l < 2; ++l) {
            int b = 5 + l * 12;
            per[l] = { F(b+0), F(b+1), F(b+2), F(b+3), F(b+4), F(b+5),
                       F(b+6), F(b+7), F(b+8), F(b+9), F(b+10), F(b+11) };
        }
        for (int l = 0; l < 2; ++l) {
            int b = 29 + l * 8;
            cr[l] = { F(b+0), F(b+1), F(b+2), F(b+3), F(b+4), F(b+5), F(b+6), F(b+7) };
        }
    }

    // ---- workspace carve (aligned 256B) ----
    char* wsp = (char*)d_ws;
    size_t cur = 0;
    auto carve = [&](size_t bytes) -> void* {
        void* p = wsp + cur;
        cur = (cur + bytes + 255) & ~(size_t)255;
        return p;
    };
    float*    queries = (float*)   carve((size_t)N_Q * DIM * 4);
    _Float16* qn16    = (_Float16*)carve((size_t)N_Q * DIM * 2);
    float*    offbuf  = (float*)   carve((size_t)N_Q * 8 * 4);
    float*    pcb     = (float*)   carve((size_t)NCAM * N_Q * 2 * 4);
    float*    valb    = (float*)   carve((size_t)NCAM * N_Q * 4);
    _Float16* samp16  = (_Float16*)carve((size_t)NCAM * N_Q * DIM * 2);
    float*    qproj   = (float*)   carve((size_t)N_Q * DIM * 4);
    float*    kvF     = (float*)   carve((size_t)NCAM * N_Q * 512 * 4); // k|v or kv
    _Float16* agg16   = (_Float16*)carve((size_t)N_Q * DIM * 2);
    _Float16 *pwq[2], *pwk[2], *pwv[2], *pwo[2], *cwq[2], *cwkv[2], *cwo[2];
    for (int l = 0; l < 2; ++l) {
        pwq[l]  = (_Float16*)carve(65536 * 2);
        pwk[l]  = (_Float16*)carve(65536 * 2);
        pwv[l]  = (_Float16*)carve(65536 * 2);
        pwo[l]  = (_Float16*)carve(65536 * 2);
        cwq[l]  = (_Float16*)carve(65536 * 2);
        cwkv[l] = (_Float16*)carve(131072 * 2);
        cwo[l]  = (_Float16*)carve(65536 * 2);
    }
    float* kF = kvF;                               // per-image k (C*N x 256)
    float* vF = kvF + (size_t)NCAM * N_Q * DIM;    // per-image v (C*N x 256)

    // ---- one-time: weight conversion/transpose, query init, projection ----
    for (int l = 0; l < 2; ++l) {
        prep_w_kernel<<<256, 256, 0, stream>>>(per[l].qW,  pwq[l],  256);
        prep_w_kernel<<<256, 256, 0, stream>>>(per[l].kW,  pwk[l],  256);
        prep_w_kernel<<<256, 256, 0, stream>>>(per[l].vW,  pwv[l],  256);
        prep_w_kernel<<<256, 256, 0, stream>>>(per[l].oW,  pwo[l],  256);
        prep_w_kernel<<<256, 256, 0, stream>>>(cr[l].qW,   cwq[l],  256);
        prep_w_kernel<<<512, 256, 0, stream>>>(cr[l].kvW,  cwkv[l], 512);
        prep_w_kernel<<<256, 256, 0, stream>>>(cr[l].oW,   cwo[l],  256);
    }
    init_q_kernel<<<N_Q, 256, 0, stream>>>(grid3d, qbase, queries);
    project_kernel<<<(NCAM * N_Q + 255) / 256, 256, 0, stream>>>(
        grid3d, intr, extr, pcb, valb);

    const int MQ  = N_Q;          // 13824, /32 = 432
    const int MKV = NCAM * N_Q;   // 82944, /32 = 2592

    // ---- per-image deformable attention layers ----
    for (int l = 0; l < 2; ++l) {
        ln_kernel<<<N_Q, 256, 0, stream>>>(queries, per[l].ng, per[l].nb,
                                           per[l].offW, per[l].offb,
                                           qn16, offbuf, 1);
        sample_kernel<<<NCAM * N_Q, 256, 0, stream>>>(feats, pcb, offbuf, samp16, 0);
        gemm_wmma_kernel<<<dim3(MQ / 32, 1),  256, 0, stream>>>(
            qn16,   pwq[l], per[l].qb, nullptr, qproj, MQ,  256);
        gemm_wmma_kernel<<<dim3(MKV / 32, 1), 256, 0, stream>>>(
            samp16, pwk[l], per[l].kb, nullptr, kF,    MKV, 256);
        gemm_wmma_kernel<<<dim3(MKV / 32, 1), 256, 0, stream>>>(
            samp16, pwv[l], per[l].vb, nullptr, vF,    MKV, 256);
        attn_per_kernel<<<N_Q, 256, 0, stream>>>(qproj, kF, vF, valb, agg16);
        gemm_wmma_kernel<<<dim3(MQ / 32, 1),  256, 0, stream>>>(
            agg16,  pwo[l], per[l].ob, queries, queries, MQ, 256);  // fused residual
    }

    // ---- cross-image attention layers ----
    for (int l = 0; l < 2; ++l) {
        ln_kernel<<<N_Q, 256, 0, stream>>>(queries, cr[l].ng, cr[l].nb,
                                           nullptr, nullptr, qn16, nullptr, 0);
        sample_kernel<<<NCAM * N_Q, 256, 0, stream>>>(feats, pcb, nullptr, samp16, 1);
        gemm_wmma_kernel<<<dim3(MQ / 32, 1),  256, 0, stream>>>(
            qn16,   cwq[l],  cr[l].qb,  nullptr, qproj, MQ,  256);
        gemm_wmma_kernel<<<dim3(MKV / 32, 2), 256, 0, stream>>>(
            samp16, cwkv[l], cr[l].kvb, nullptr, kvF,   MKV, 512);
        attn_cross_kernel<<<N_Q, 256, 0, stream>>>(qproj, kvF, valb, agg16);
        gemm_wmma_kernel<<<dim3(MQ / 32, 1),  256, 0, stream>>>(
            agg16,  cwo[l],  cr[l].ob,  queries, queries, MQ, 256);
    }

    // ---- final pooling to the three BEV planes ----
    reduce_kernel<<<dim3(SS, SS, 3), 256, 0, stream>>>(queries, (float*)d_out);
}
